// VectorizedLIFLayer_57535381897773
// MI455X (gfx1250) — matrix operand order repaired and checked by reading, not compiled, not run on hardware
//
#include <hip/hip_runtime.h>
#include <hip/hip_bf16.h>
#include <math.h>

typedef __attribute__((ext_vector_type(16))) __bf16 v16bf;
typedef __attribute__((ext_vector_type(8)))  float  v8f;
typedef __attribute__((ext_vector_type(4)))  float  v4f;

#define B_SZ  64
#define T_SZ  1024
#define F_IN  512
#define F_OUT 512
#define M_TOT (B_SZ * T_SZ)   // 65536 flattened (b,t) rows

#define WMMA_BF16(a, b, c) \
  __builtin_amdgcn_wmma_f32_16x16x32_bf16(false, (a), false, (b), (short)0, (c), false, false)

// ---------------------------------------------------------------------------
// Kernel 1: transpose + convert W [K=512][N=512] f32  ->  Wt [N][K] bf16
// (512 KB, stays L2/WGP$-resident; gives each WMMA lane a contiguous 32B load)
// ---------------------------------------------------------------------------
__global__ __launch_bounds__(256) void wt_transpose_kernel(
    const float* __restrict__ W, __bf16* __restrict__ Wt) {
  int idx = blockIdx.x * 256 + threadIdx.x;   // 0 .. 512*512-1
  int n = idx & (F_OUT - 1);
  int k = idx >> 9;
  Wt[(size_t)n * F_IN + k] = (__bf16)W[(size_t)k * F_OUT + n];
}

// ---------------------------------------------------------------------------
// Kernel 2: WMMA GEMM  currents[m,n] = sum_k spikes[m,k]*W[k,n] + b[n]
// Wave tile: 32(M) x 64(N) -> 8 v_wmma_f32_16x16x32_bf16 per k-step.
// K loop fully unrolled (16 steps): prefetch guard constant-folds, compiler
// renames the ping-pong registers (no v_mov rotation), loads for step i+1
// stay in flight across step i's 8 WMMAs via staged s_wait_loadcnt.
// Block tile: 64(M) x 256(N), 8 waves. Grid: (M_TOT/64) x (F_OUT/256).
// ---------------------------------------------------------------------------
__global__ __launch_bounds__(256) void gemm_wmma_kernel(
    const float*  __restrict__ spikes,  // [M_TOT][F_IN] f32
    const __bf16* __restrict__ Wt,      // [F_OUT][F_IN] bf16 (transposed)
    const float*  __restrict__ bias,    // [F_OUT]
    float*        __restrict__ cur)     // [M_TOT][F_OUT] f32
{
  const int lane = threadIdx.x & 31;
  const int wave = threadIdx.x >> 5;
  const int mt   = blockIdx.x * 64 + (wave & 1) * 32;   // 32 rows / wave
  const int nb   = blockIdx.y * 256 + (wave >> 1) * 64; // 64 cols / wave
  const int lm   = lane & 15;
  const int hi   = lane >> 4;            // 0: lanes 0-15, 1: lanes 16-31

  // A (16x32 bf16): lane holds row m = lm; elements 0-15 <- k+16*hi+{0..15}.
  // Two row-tiles (mt, mt+16).
  const float* arow0 = spikes + (size_t)(mt + lm) * F_IN + (hi << 4);
  const float* arow1 = arow0 + (size_t)16 * F_IN;
  // B (32x16 bf16): lane holds col n = nb+16j+lm; k offset 16*hi, 16 contiguous.
  const __bf16* bp0 = Wt + (size_t)(nb +  0 + lm) * F_IN + (hi << 4);
  const __bf16* bp1 = Wt + (size_t)(nb + 16 + lm) * F_IN + (hi << 4);
  const __bf16* bp2 = Wt + (size_t)(nb + 32 + lm) * F_IN + (hi << 4);
  const __bf16* bp3 = Wt + (size_t)(nb + 48 + lm) * F_IN + (hi << 4);

  v8f accA[4] = {{0}, {0}, {0}, {0}};   // m-tile 0, n-tiles 0..3
  v8f accB[4] = {{0}, {0}, {0}, {0}};   // m-tile 1, n-tiles 0..3

  // ---- prologue: load + convert k-tile 0 ----
  v16bf av0, av1, bc0, bc1, bc2, bc3;
  {
    v8f a00 = *(const v8f*)(arow0);
    v8f a01 = *(const v8f*)(arow0 + 8);
    v8f a10 = *(const v8f*)(arow1);
    v8f a11 = *(const v8f*)(arow1 + 8);
    bc0 = *(const v16bf*)(bp0);
    bc1 = *(const v16bf*)(bp1);
    bc2 = *(const v16bf*)(bp2);
    bc3 = *(const v16bf*)(bp3);
#pragma unroll
    for (int e = 0; e < 8; ++e) {
      av0[e] = (__bf16)a00[e]; av0[e + 8] = (__bf16)a01[e];
      av1[e] = (__bf16)a10[e]; av1[e + 8] = (__bf16)a11[e];
    }
  }

  // ---- fully-unrolled pipelined main loop: 16 k-steps of 32 ----
#pragma unroll
  for (int kt = 32; kt <= F_IN; kt += 32) {
    v8f na00, na01, na10, na11;
    v16bf nb0, nb1, nb2, nb3;
    const bool pf = (kt < F_IN);        // constant per unrolled iteration
    if (pf) {   // issue next tile's loads; stay in flight during the WMMAs
      na00 = *(const v8f*)(arow0 + kt);
      na01 = *(const v8f*)(arow0 + kt + 8);
      na10 = *(const v8f*)(arow1 + kt);
      na11 = *(const v8f*)(arow1 + kt + 8);
      nb0  = *(const v16bf*)(bp0 + kt);
      nb1  = *(const v16bf*)(bp1 + kt);
      nb2  = *(const v16bf*)(bp2 + kt);
      nb3  = *(const v16bf*)(bp3 + kt);
    }

    // 8 independent WMMAs on the already-resident tile
    accA[0] = WMMA_BF16(av0, bc0, accA[0]);
    accB[0] = WMMA_BF16(av1, bc0, accB[0]);
    accA[1] = WMMA_BF16(av0, bc1, accA[1]);
    accB[1] = WMMA_BF16(av1, bc1, accB[1]);
    accA[2] = WMMA_BF16(av0, bc2, accA[2]);
    accB[2] = WMMA_BF16(av1, bc2, accB[2]);
    accA[3] = WMMA_BF16(av0, bc3, accA[3]);
    accB[3] = WMMA_BF16(av1, bc3, accB[3]);

    if (pf) {   // stage next tile (loads have had the 8 WMMAs to complete)
#pragma unroll
      for (int e = 0; e < 8; ++e) {
        av0[e] = (__bf16)na00[e]; av0[e + 8] = (__bf16)na01[e];
        av1[e] = (__bf16)na10[e]; av1[e + 8] = (__bf16)na11[e];
      }
      bc0 = nb0; bc1 = nb1; bc2 = nb2; bc3 = nb3;
    }
  }

  // ---- epilogue: bias + store ----
  const float bi0 = bias[nb +  0 + lm];
  const float bi1 = bias[nb + 16 + lm];
  const float bi2 = bias[nb + 32 + lm];
  const float bi3 = bias[nb + 48 + lm];

  // C/D layout: VGPR i -> row (tile_base + i + 8*hi), col nb + 16j + lm
  float* o0 = cur + (size_t)(mt + (hi << 3)) * F_OUT + nb + lm;
  float* o1 = o0 + (size_t)16 * F_OUT;
#pragma unroll
  for (int i = 0; i < 8; ++i) {
    size_t r = (size_t)i * F_OUT;
    o0[r +  0] = accA[0][i] + bi0;
    o0[r + 16] = accA[1][i] + bi1;
    o0[r + 32] = accA[2][i] + bi2;
    o0[r + 48] = accA[3][i] + bi3;
    o1[r +  0] = accB[0][i] + bi0;
    o1[r + 16] = accB[1][i] + bi1;
    o1[r + 32] = accB[2][i] + bi2;
    o1[r + 48] = accB[3][i] + bi3;
  }
}

// ---------------------------------------------------------------------------
// Kernel 3: LIF scan. One thread per (b,f) sequence (32768 threads = 1024
// waves), coalesced across f every time step; currents is L2-resident.
// alpha = exp(-1/0.02) = exp(-50)
// ---------------------------------------------------------------------------
__global__ __launch_bounds__(256) void lif_scan_kernel(
    const float* __restrict__ cur, float* __restrict__ out) {
  int idx = blockIdx.x * 256 + threadIdx.x;      // 0 .. B*F-1
  int b = idx >> 9;
  int f = idx & (F_OUT - 1);
  size_t base = (size_t)b * T_SZ * F_OUT + f;

  const float alpha    = 1.9287498479639178e-22f;   // exp(-50), f32
  const float om_alpha = 1.0f - alpha;
  float v = 0.0f;
#pragma unroll 4
  for (int t = 0; t < T_SZ; ++t) {
    float c = cur[base + (size_t)t * F_OUT];
    v = alpha * v + om_alpha * c;
    float bx = 4.0f * (v - 0.5f);                    // beta * (v - threshold)
    float s  = 0.5f * bx / (1.0f + fabsf(bx)) + 0.5f;
    v = v * (1.0f - s);                               // RESET_POTENTIAL = 0
    out[base + (size_t)t * F_OUT] = s;
  }
}

// ---------------------------------------------------------------------------
// Kernel 4: LayerNorm over F. One wave32 per (b,t) row of 512 floats;
// shuffle-tree reduction, population variance (jnp.var), eps = 1e-6.
// ---------------------------------------------------------------------------
__global__ __launch_bounds__(256) void layernorm_kernel(
    float* __restrict__ out,
    const float* __restrict__ gamma, const float* __restrict__ beta) {
  const int lane = threadIdx.x & 31;
  const int wave = threadIdx.x >> 5;
  const size_t row = (size_t)blockIdx.x * 8 + wave;   // 65536 rows
  float* p = out + row * F_OUT;

  v4f x[4];
  float s = 0.0f, ss = 0.0f;
#pragma unroll
  for (int i = 0; i < 4; ++i) {
    int f0 = ((i << 5) + lane) << 2;                  // 128 float4s / row
    x[i] = *(const v4f*)(p + f0);
#pragma unroll
    for (int c = 0; c < 4; ++c) { float v = x[i][c]; s += v; ss += v * v; }
  }
#pragma unroll
  for (int m = 16; m >= 1; m >>= 1) {
    s  += __shfl_xor(s,  m, 32);
    ss += __shfl_xor(ss, m, 32);
  }
  const float inv  = 1.0f / (float)F_OUT;
  float mean = s * inv;
  float var  = ss * inv - mean * mean;
  float rstd = rsqrtf(var + 1e-6f);

#pragma unroll
  for (int i = 0; i < 4; ++i) {
    int f0 = ((i << 5) + lane) << 2;
    v4f g  = *(const v4f*)(gamma + f0);
    v4f bt = *(const v4f*)(beta  + f0);
    v4f y;
#pragma unroll
    for (int c = 0; c < 4; ++c) y[c] = (x[i][c] - mean) * rstd * g[c] + bt[c];
    *(v4f*)(p + f0) = y;
  }
}

// ---------------------------------------------------------------------------
extern "C" void kernel_launch(void* const* d_in, const int* in_sizes, int n_in,
                              void* d_out, int out_size, void* d_ws, size_t ws_size,
                              hipStream_t stream) {
  const float* spikes = (const float*)d_in[0];   // [64,1024,512]
  const float* W      = (const float*)d_in[1];   // [512,512]
  const float* bias   = (const float*)d_in[2];   // [512]
  const float* gamma  = (const float*)d_in[3];   // [512]
  const float* beta   = (const float*)d_in[4];   // [512]
  float* out = (float*)d_out;                    // [64,1024,512] f32

  // ws layout: [Wt bf16 512KB][currents f32 128MB] => ~134.7 MB total
  __bf16* Wt  = (__bf16*)d_ws;
  float*  cur = (float*)((char*)d_ws + (size_t)F_OUT * F_IN * sizeof(__bf16));

  wt_transpose_kernel<<<(F_IN * F_OUT) / 256, 256, 0, stream>>>(W, Wt);

  dim3 ggrid(M_TOT / 64, F_OUT / 256);
  gemm_wmma_kernel<<<ggrid, 256, 0, stream>>>(spikes, Wt, bias, cur);

  lif_scan_kernel<<<(B_SZ * F_OUT) / 256, 256, 0, stream>>>(cur, out);
  layernorm_kernel<<<M_TOT / 8, 256, 0, stream>>>(out, gamma, beta);
}
